// LSTMCell_64879775973867
// MI455X (gfx1250) — compile-verified, matrix operand order checked by reading, not run on hardware
//
#include <hip/hip_runtime.h>

// ---------------------------------------------------------------------------
// 4-layer stacked LSTM cell, B=4096, E=H=1024, K=E+H=2048, N=4H=4096 per layer.
// bf16 WMMA (v_wmma_f32_16x16x32_bf16) with fp32 accumulate; gate-interleaved
// N tiling so activations + cell update fuse in registers.
// ---------------------------------------------------------------------------

typedef __attribute__((ext_vector_type(16))) __bf16 v16bf;
typedef __attribute__((ext_vector_type(2)))  __bf16 v2bf;
typedef __attribute__((ext_vector_type(8)))  float  v8f;

#define B_ROWS 4096
#define HID    1024
#define KDIM   2048          // E + H
#define GATES  4
#define NCOLS  (GATES * HID) // 4096

#define TM 128               // rows per block
#define TH 32                // hidden units per block (x4 gates = 128 cols)
#define TK 32                // K per WMMA step
#define LDS_STRIDE 40        // 32 data + 8 pad ushorts = 80B: 16B-aligned rows,
                             // conflict-free b128 fragment loads (stride 20 dwords)

// packed fp32x2 -> bf16x2 (compiler selects v_cvt_pk_bf16_f32)
__device__ __forceinline__ unsigned int pack2(float a, float b) {
    v2bf t;
    t[0] = (__bf16)a;
    t[1] = (__bf16)b;
    return __builtin_bit_cast(unsigned int, t);
}

__device__ __forceinline__ float sigmoidf_(float x) {
    return 1.0f / (1.0f + __expf(-x));
}

union Frag16 { v16bf v; int4 q[2]; };

__global__ __launch_bounds__(256)
void lstm_layer_wmma(const float* __restrict__ xin,   // [B, HID] layer input
                     const float* __restrict__ hin,   // [B, HID] s_h slice
                     const float* __restrict__ cin,   // [B, HID] s_c slice
                     const float* __restrict__ W,     // [KDIM, NCOLS]
                     const float* __restrict__ bias,  // [NCOLS]
                     float* __restrict__ hout,        // [B, HID]
                     float* __restrict__ cout,        // [B, HID]
                     float* __restrict__ finalOut)    // [B, HID] or nullptr
{
    __shared__ unsigned short Alds[TM * LDS_STRIDE];          // [row][k] bf16
    __shared__ unsigned short Blds[(TH * GATES) * LDS_STRIDE];// [gatecol][k] bf16 (W^T)

    const int tid   = threadIdx.x;
    const int lane  = tid & 31;
    const int wave  = tid >> 5;     // 0..7
    const int waveM = wave >> 1;    // 0..3 : 32-row strip
    const int waveN = wave & 1;     // 0..1 : 16-hidden-unit strip
    const int lm    = lane & 15;
    const int lh    = lane >> 4;    // 0/1
    const int koff  = lh * 8;       // A/B fragment K base (16-bit WMMA layout)

    const int m0 = blockIdx.y * TM; // row block
    const int h0 = blockIdx.x * TH; // hidden-unit block

    // global -> LDS work split (256 threads)
    const int ar  = tid >> 1;            // A: row 0..127
    const int akq = (tid & 1) * 16;      // A: k sub-offset 0/16
    const int bkp = (tid & 15) * 2;      // B: even K row 0..30 (handles k, k+1)
    const int bc0 = (tid >> 4) * 8;      // B: gate-col group 0..120
    const int bg  = bc0 >> 5;            // gate 0..3
    const int bj  = bc0 & 31;            // 0/8/16/24 within gate

    v8f acc[2][GATES];
#pragma unroll
    for (int mt = 0; mt < 2; ++mt)
#pragma unroll
        for (int g = 0; g < GATES; ++g)
            acc[mt][g] = (v8f){0.f, 0.f, 0.f, 0.f, 0.f, 0.f, 0.f, 0.f};

    for (int k0 = 0; k0 < KDIM; k0 += TK) {
        // ---- A source: concat(out, h) handled by pointer switch at K=HID ----
        const float* Asrc = (k0 < HID) ? xin : hin;
        const int    kk   = (k0 < HID) ? k0 : (k0 - HID);

        // ---- stage A tile: 128 x 32 fp32 -> bf16 LDS (two b128 stores) ----
        {
            const float4* p = reinterpret_cast<const float4*>(
                Asrc + (size_t)(m0 + ar) * HID + kk + akq);
            float4 f0 = p[0], f1 = p[1], f2 = p[2], f3 = p[3];
            int4 s0 = make_int4((int)pack2(f0.x, f0.y), (int)pack2(f0.z, f0.w),
                                (int)pack2(f1.x, f1.y), (int)pack2(f1.z, f1.w));
            int4 s1 = make_int4((int)pack2(f2.x, f2.y), (int)pack2(f2.z, f2.w),
                                (int)pack2(f3.x, f3.y), (int)pack2(f3.z, f3.w));
            int4* d = reinterpret_cast<int4*>(&Alds[ar * LDS_STRIDE + akq]);
            d[0] = s0;
            d[1] = s1;
        }

        // ---- stage B tile transposed: each thread owns K rows (bkp, bkp+1) x 8
        //      cols; packs (k,k+1) per column -> one b32 store per column ----
        {
            const float* prow = W + (size_t)(k0 + bkp) * NCOLS
                                  + (size_t)bg * HID + h0 + bj;
            const float4* p0 = reinterpret_cast<const float4*>(prow);
            const float4* p1 = reinterpret_cast<const float4*>(prow + NCOLS);
            union { float4 f[2]; float s[8]; } w0, w1;
            w0.f[0] = p0[0]; w0.f[1] = p0[1];
            w1.f[0] = p1[0]; w1.f[1] = p1[1];
#pragma unroll
            for (int j = 0; j < 8; ++j) {
                *reinterpret_cast<unsigned int*>(
                    &Blds[(bc0 + j) * LDS_STRIDE + bkp]) = pack2(w0.s[j], w1.s[j]);
            }
        }

        // prefetch next K-tile of W into cache (global_prefetch_b8)
        if (k0 + TK < KDIM) {
            __builtin_prefetch(
                W + (size_t)(k0 + TK + bkp) * NCOLS + (size_t)bg * HID + h0 + bj,
                0, 1);
        }

        __syncthreads();

        // ---- fragments: A lane<16 holds K{0..7,16..23}, lane>=16 K{8..15,24..31};
        //      B mirrors A on the (transposed) gate-column index ----
        Frag16 afr[2], bfr[GATES];
#pragma unroll
        for (int mt = 0; mt < 2; ++mt) {
            const unsigned short* pa =
                &Alds[(waveM * 32 + mt * 16 + lm) * LDS_STRIDE + koff];
            afr[mt].q[0] = *reinterpret_cast<const int4*>(pa);
            afr[mt].q[1] = *reinterpret_cast<const int4*>(pa + 16);
        }
#pragma unroll
        for (int g = 0; g < GATES; ++g) {
            const unsigned short* pb =
                &Blds[(g * TH + waveN * 16 + lm) * LDS_STRIDE + koff];
            bfr[g].q[0] = *reinterpret_cast<const int4*>(pb);
            bfr[g].q[1] = *reinterpret_cast<const int4*>(pb + 16);
        }

#pragma unroll
        for (int mt = 0; mt < 2; ++mt)
#pragma unroll
            for (int g = 0; g < GATES; ++g)
                acc[mt][g] = __builtin_amdgcn_wmma_f32_16x16x32_bf16(
                    false, afr[mt].v, false, bfr[g].v,
                    (short)0, acc[mt][g], false, false);

        __syncthreads();
    }

    // ---- fused epilogue: this wave owns i,f,o,g for the same 16 hidden units ----
    const int col = h0 + waveN * 16 + lm;   // hidden unit index
    float bz[GATES];
#pragma unroll
    for (int g = 0; g < GATES; ++g) bz[g] = bias[g * HID + col];

#pragma unroll
    for (int mt = 0; mt < 2; ++mt) {
#pragma unroll
        for (int r = 0; r < 8; ++r) {
            const int row = m0 + waveM * 32 + mt * 16 + lh * 8 + r; // C/D layout
            const size_t idx = (size_t)row * HID + col;
            const float ig = sigmoidf_(acc[mt][0][r] + bz[0]);
            const float fg = sigmoidf_(acc[mt][1][r] + bz[1]);
            const float og = sigmoidf_(acc[mt][2][r] + bz[2]);
            const float gg = tanhf(acc[mt][3][r] + bz[3]);
            const float cn = cin[idx] * fg + gg * ig;
            const float hn = tanhf(cn) * og;
            cout[idx] = cn;
            hout[idx] = hn;
            if (finalOut) finalOut[idx] = hn;
        }
    }
}

extern "C" void kernel_launch(void* const* d_in, const int* in_sizes, int n_in,
                              void* d_out, int out_size, void* d_ws, size_t ws_size,
                              hipStream_t stream) {
    const float* x   = (const float*)d_in[0];   // [B, E]
    const float* s_h = (const float*)d_in[1];   // [4, B, H]
    const float* s_c = (const float*)d_in[2];   // [4, B, H]
    const float* W   = (const float*)d_in[3];   // [4, KDIM, NCOLS]
    const float* b   = (const float*)d_in[4];   // [4, NCOLS]

    float* out  = (float*)d_out;                          // [B, H]
    float* hnew = out + (size_t)B_ROWS * HID;             // [4, B, H]
    float* cnew = hnew + (size_t)4 * B_ROWS * HID;        // [4, B, H]

    const dim3 grid(HID / TH, B_ROWS / TM);   // (32, 32)
    const dim3 block(256);

    for (int l = 0; l < 4; ++l) {
        const float* layerIn = (l == 0) ? x : (hnew + (size_t)(l - 1) * B_ROWS * HID);
        lstm_layer_wmma<<<grid, block, 0, stream>>>(
            layerIn,
            s_h + (size_t)l * B_ROWS * HID,
            s_c + (size_t)l * B_ROWS * HID,
            W   + (size_t)l * KDIM * NCOLS,
            b   + (size_t)l * NCOLS,
            hnew + (size_t)l * B_ROWS * HID,
            cnew + (size_t)l * B_ROWS * HID,
            (l == 3) ? out : nullptr);
    }
}
